// minerva2_62912680951919
// MI455X (gfx1250) — compile-verified, matrix operand order checked by reading, not run on hardware
//
#include <hip/hip_runtime.h>
#include <hip/hip_bf16.h>

// Problem sizes (fixed by the reference setup_inputs()).
constexpr int N_ = 8192;   // feature rows
constexpr int D_ = 16384;  // exemplar rows
constexpr int H_ = 1024;   // feature dim
constexpr int C_ = 512;    // class-rep dim
constexpr float EPS_ = 1e-12f;

typedef __attribute__((ext_vector_type(16))) _Float16 v16h;
typedef __attribute__((ext_vector_type(8)))  _Float16 v8h;
typedef __attribute__((ext_vector_type(8)))  float    v8f;

// ---------------------------------------------------------------------------
// WMMA helper: D = A(16x32 f16) x B(32x16 f16) + C(16x16 f32)
// ---------------------------------------------------------------------------
__device__ __forceinline__ v8f wmma_f16(v16h a, v16h b, v8f c) {
  return __builtin_amdgcn_wmma_f32_16x16x32_f16(
      /*neg_a=*/false, a, /*neg_b=*/false, b,
      /*c_mod=*/(short)0, c, /*reuse_a=*/false, /*reuse_b=*/false);
}

// ---------------------------------------------------------------------------
// Fragment loaders.
// ISA 16-bit A/B layout (16x32, M or N = lane%16):
//   lane L, g = L/16 holds K-chunks [8g, 8g+8) in elements 0..7 (VGPR0..3)
//   and [16+8g, 16+8g+8) in elements 8..15 (VGPR4..7).
// Both chunks are 8 contiguous f16 = one 16-byte load each.
// ---------------------------------------------------------------------------
__device__ __forceinline__ v16h load_frag_f16(const _Float16* __restrict__ base,
                                              int row_stride, int row0, int k,
                                              int lane) {
  const int r = row0 + (lane & 15);
  const int g = lane >> 4;
  const _Float16* p = base + (size_t)r * row_stride + k + 8 * g;
  v8h lo = *(const v8h*)(p);
  v8h hi = *(const v8h*)(p + 16);
  v16h out;
#pragma unroll
  for (int i = 0; i < 8; ++i) { out[i] = lo[i]; out[i + 8] = hi[i]; }
  return out;
}

// Same layout, but source matrix is f32: load and convert to f16 in registers.
__device__ __forceinline__ v16h load_frag_f32cvt(const float* __restrict__ base,
                                                 int row_stride, int row0, int k,
                                                 int lane) {
  const int r = row0 + (lane & 15);
  const int g = lane >> 4;
  const float* p = base + (size_t)r * row_stride + k + 8 * g;
  float4 l0 = *(const float4*)(p);
  float4 l1 = *(const float4*)(p + 4);
  float4 h0 = *(const float4*)(p + 16);
  float4 h1 = *(const float4*)(p + 20);
  v16h out;
  out[0]  = (_Float16)l0.x; out[1]  = (_Float16)l0.y;
  out[2]  = (_Float16)l0.z; out[3]  = (_Float16)l0.w;
  out[4]  = (_Float16)l1.x; out[5]  = (_Float16)l1.y;
  out[6]  = (_Float16)l1.z; out[7]  = (_Float16)l1.w;
  out[8]  = (_Float16)h0.x; out[9]  = (_Float16)h0.y;
  out[10] = (_Float16)h0.z; out[11] = (_Float16)h0.w;
  out[12] = (_Float16)h1.x; out[13] = (_Float16)h1.y;
  out[14] = (_Float16)h1.z; out[15] = (_Float16)h1.w;
  return out;
}

// C/D 16x16 f32 layout: element r of lane L is (M = r + 8*(L/16), N = L%16).
__device__ __forceinline__ void store_tile_cube(float* __restrict__ out,
                                                int out_stride, int row0,
                                                int col0, v8f c, int lane) {
  const int col = col0 + (lane & 15);
  const int rbase = row0 + ((lane >> 4) * 8);
#pragma unroll
  for (int r = 0; r < 8; ++r) {
    float v = c[r];
    out[(size_t)(rbase + r) * out_stride + col] = v * v * v;  // sign(s)|s|^3
  }
}

__device__ __forceinline__ void store_tile(float* __restrict__ out,
                                           int out_stride, int row0, int col0,
                                           v8f c, int lane) {
  const int col = col0 + (lane & 15);
  const int rbase = row0 + ((lane >> 4) * 8);
#pragma unroll
  for (int r = 0; r < 8; ++r) {
    out[(size_t)(rbase + r) * out_stride + col] = c[r];
  }
}

// ---------------------------------------------------------------------------
// K1: per-row L2 normalize (f32 -> f16).  One 256-thread block per row.
// ---------------------------------------------------------------------------
__global__ __launch_bounds__(256) void norm_rows_f16(
    const float* __restrict__ src, _Float16* __restrict__ dst, int H) {
  const int row = blockIdx.x;
  const float* s = src + (size_t)row * H;
  float ss = 0.0f;
  for (int i = threadIdx.x; i < H; i += 256) {
    float v = s[i];
    ss += v * v;
  }
#pragma unroll
  for (int off = 16; off > 0; off >>= 1) ss += __shfl_down(ss, off, 32);
  __shared__ float red[8];
  __shared__ float inv_s;
  if ((threadIdx.x & 31) == 0) red[threadIdx.x >> 5] = ss;
  __syncthreads();
  if (threadIdx.x == 0) {
    float t = 0.0f;
#pragma unroll
    for (int i = 0; i < 8; ++i) t += red[i];
    inv_s = 1.0f / fmaxf(sqrtf(t), EPS_);
  }
  __syncthreads();
  const float inv = inv_s;
  _Float16* d = dst + (size_t)row * H;
  for (int i = threadIdx.x; i < H; i += 256) d[i] = (_Float16)(s[i] * inv);
}

// ---------------------------------------------------------------------------
// K2: ex_class_reps [D,C] f32 -> transposed [C,D] f16 (B-fragments contiguous)
// ---------------------------------------------------------------------------
__global__ __launch_bounds__(256) void cvt_transpose_rep(
    const float* __restrict__ src, _Float16* __restrict__ dst) {
  size_t i = (size_t)blockIdx.x * blockDim.x + threadIdx.x;
  if (i < (size_t)D_ * C_) {
    int d = (int)(i / C_);
    int c = (int)(i % C_);
    dst[(size_t)c * D_ + d] = (_Float16)src[i];
  }
}

// ---------------------------------------------------------------------------
// K3: GEMM1 + cube.  s = featH @ exH^T, a_un = s^3  (f32 out, [N,D]).
// Block: 256 threads = 8 waves arranged 4(M) x 2(D); block tile 128 x 128;
// wave tile 32x64 via 2x4 v_wmma_f32_16x16x32_f16 (1.5 b128 loads / WMMA).
// Both f16 operands (16 + 32 MB) are L2-resident, so HBM cost is the a-write.
// Prefetch is unconditional: gfx1250 prefetch is speculative, OOB addresses
// are dropped silently, and the guard was forcing a branchy inner loop.
// ---------------------------------------------------------------------------
__global__ __launch_bounds__(256) void gemm1_cube(
    const _Float16* __restrict__ featH, const _Float16* __restrict__ exH,
    float* __restrict__ aOut) {
  const int lane = threadIdx.x;       // 0..31
  const int wave = threadIdx.y;       // 0..7
  const int m0 = blockIdx.y * 128 + (wave & 3) * 32;
  const int d0 = blockIdx.x * 128 + (wave >> 2) * 64;

  v8f acc[2][4] = {};
  for (int k = 0; k < H_; k += 32) {
    __builtin_prefetch(featH + (size_t)(m0 + (lane & 15)) * H_ + k + 32, 0, 3);
    __builtin_prefetch(exH + (size_t)(d0 + (lane & 15)) * H_ + k + 32, 0, 3);
    v16h a[2], b[4];
#pragma unroll
    for (int i = 0; i < 2; ++i)
      a[i] = load_frag_f16(featH, H_, m0 + 16 * i, k, lane);
#pragma unroll
    for (int j = 0; j < 4; ++j)
      b[j] = load_frag_f16(exH, H_, d0 + 16 * j, k, lane);
#pragma unroll
    for (int i = 0; i < 2; ++i)
#pragma unroll
      for (int j = 0; j < 4; ++j)
        acc[i][j] = wmma_f16(a[i], b[j], acc[i][j]);
  }
#pragma unroll
  for (int i = 0; i < 2; ++i)
#pragma unroll
    for (int j = 0; j < 4; ++j)
      store_tile_cube(aOut, D_, m0 + 16 * i, d0 + 16 * j, acc[i][j], lane);
}

// ---------------------------------------------------------------------------
// K4: per-row L1 normalize of a (in place).  One block per row; the second
// pass re-reads the 64 KB row straight out of L2.
// ---------------------------------------------------------------------------
__global__ __launch_bounds__(256) void l1_scale_rows(float* __restrict__ a) {
  const int row = blockIdx.x;
  float* r = a + (size_t)row * D_;
  float s = 0.0f;
  for (int i = threadIdx.x; i < D_ / 4; i += 256) {
    float4 v = ((const float4*)r)[i];
    s += fabsf(v.x) + fabsf(v.y) + fabsf(v.z) + fabsf(v.w);
  }
#pragma unroll
  for (int off = 16; off > 0; off >>= 1) s += __shfl_down(s, off, 32);
  __shared__ float red[8];
  __shared__ float inv_s;
  if ((threadIdx.x & 31) == 0) red[threadIdx.x >> 5] = s;
  __syncthreads();
  if (threadIdx.x == 0) {
    float t = 0.0f;
#pragma unroll
    for (int i = 0; i < 8; ++i) t += red[i];
    inv_s = 1.0f / fmaxf(t, EPS_);
  }
  __syncthreads();
  const float inv = inv_s;
  for (int i = threadIdx.x; i < D_ / 4; i += 256) {
    float4 v = ((float4*)r)[i];
    v.x *= inv; v.y *= inv; v.z *= inv; v.w *= inv;
    ((float4*)r)[i] = v;
  }
}

// ---------------------------------------------------------------------------
// K5: GEMM2.  echo = a @ reps  ([N,D] f32 x [D,C] -> [N,C] f32).
// One block covers the FULL C extent: block tile 32(M) x 512(C), 8 waves of
// 32x64 each.  Every element of the 536 MB a matrix is read exactly once from
// HBM; the 16 MB f16 repT operand is L2-resident across all 256 blocks.
// A-fragments convert f32 -> f16 in registers (co-executes with WMMA, which
// is tracked as a TRANS op on CDNA5).
// ---------------------------------------------------------------------------
__global__ __launch_bounds__(256) void gemm2_echo(
    const float* __restrict__ a, const _Float16* __restrict__ repT,
    float* __restrict__ echo) {
  const int lane = threadIdx.x;
  const int wave = threadIdx.y;
  const int m0 = blockIdx.x * 32;
  const int c0 = wave * 64;

  v8f acc[2][4] = {};
  for (int k = 0; k < D_; k += 32) {
    // a is the one HBM-resident stream: speculative prefetch, no guard.
    __builtin_prefetch(a + (size_t)(m0 + (lane & 15)) * D_ + k + 32, 0, 3);
    v16h af[2], b[4];
#pragma unroll
    for (int i = 0; i < 2; ++i)
      af[i] = load_frag_f32cvt(a, D_, m0 + 16 * i, k, lane);
#pragma unroll
    for (int j = 0; j < 4; ++j)
      b[j] = load_frag_f16(repT, D_, c0 + 16 * j, k, lane);
#pragma unroll
    for (int i = 0; i < 2; ++i)
#pragma unroll
      for (int j = 0; j < 4; ++j)
        acc[i][j] = wmma_f16(af[i], b[j], acc[i][j]);
  }
#pragma unroll
  for (int i = 0; i < 2; ++i)
#pragma unroll
    for (int j = 0; j < 4; ++j)
      store_tile(echo, C_, m0 + 16 * i, c0 + 16 * j, acc[i][j], lane);
}

// ---------------------------------------------------------------------------
// Host-side launch.  d_in: {features [N,H] f32, ex_features [D,H] f32,
// ex_class_reps [D,C] f32}.  d_out: echo [N,C] f32 ++ a [N,D] f32.
// Workspace: featH (16 MB) | exH (32 MB) | repT (16 MB), all f16.
// ---------------------------------------------------------------------------
extern "C" void kernel_launch(void* const* d_in, const int* in_sizes, int n_in,
                              void* d_out, int out_size, void* d_ws,
                              size_t ws_size, hipStream_t stream) {
  const float* features = (const float*)d_in[0];
  const float* ex_feat  = (const float*)d_in[1];
  const float* ex_reps  = (const float*)d_in[2];

  _Float16* featH = (_Float16*)d_ws;                 // N*H f16
  _Float16* exH   = featH + (size_t)N_ * H_;         // D*H f16
  _Float16* repT  = exH + (size_t)D_ * H_;           // C*D f16

  float* echo = (float*)d_out;                       // N*C
  float* aMat = echo + (size_t)N_ * C_;              // N*D

  norm_rows_f16<<<N_, 256, 0, stream>>>(features, featH, H_);
  norm_rows_f16<<<D_, 256, 0, stream>>>(ex_feat, exH, H_);

  {
    size_t total = (size_t)D_ * C_;
    int blocks = (int)((total + 255) / 256);
    cvt_transpose_rep<<<blocks, 256, 0, stream>>>(ex_reps, repT);
  }

  dim3 blk(32, 8);
  gemm1_cube<<<dim3(D_ / 128, N_ / 128), blk, 0, stream>>>(featH, exH, aMat);
  l1_scale_rows<<<N_, 256, 0, stream>>>(aMat);
  gemm2_echo<<<N_ / 32, blk, 0, stream>>>(aMat, repT, echo);
}
